// OuterProductMean_21766894256707
// MI455X (gfx1250) — compile-verified, matrix operand order checked by reading, not run on hardware
//
#include <hip/hip_runtime.h>
#include <hip/hip_bf16.h>

// Sizes from the reference
#define S_DIM   128
#define I_DIM   512
#define CIN     64
#define C_DIM   32
#define D_DIM   32
#define CZ      128
#define CD      (C_DIM * D_DIM)   // 1024

// Fused-kernel tiling
#define IT      16                // i-tile per block
#define JT      16                // j-tile per block
#define CC      8                 // c-chunk (K-chunk of GEMM2 = CC*32 = 256)
#define NCHUNK  (C_DIM / CC)      // 4

#define ZROWS   (IT * JT)         // 256 (i,j) pairs
#define ZCOLS   (CC * D_DIM)      // 256 cd per chunk
#define ZS      (ZCOLS + 16)      // padded LDS stride (32B aligned)
#define BROWS   (JT * D_DIM)      // 512 (j,d) rows of the staged b-tile
#define BS      (S_DIM + 16)      // padded LDS stride (32B aligned)

typedef _Float16 v16h __attribute__((ext_vector_type(16)));
typedef float    v8f  __attribute__((ext_vector_type(8)));

// ---------------------------------------------------------------------------
// Kernel 0: transpose + convert o_weight [1024,128] f32 -> oWT [128][1024] f16
// ---------------------------------------------------------------------------
__global__ void k_prep_ow(const float* __restrict__ oW, _Float16* __restrict__ oWT) {
  int idx = blockIdx.x * 256 + threadIdx.x;      // 0 .. 131071
  if (idx < CD * CZ) {
    int k = idx >> 7;      // cd index (idx = k*128 + z)
    int z = idx & (CZ - 1);
    oWT[(size_t)z * CD + k] = (_Float16)oW[idx];
  }
}

// ---------------------------------------------------------------------------
// Kernel 1: LayerNorm(64) + A/B projections (64->32 each).
// One wave per (s,i) row. K-major (S-contiguous) f16 outputs:
//   aT[i][c][s] (mask and 1/S folded in),  bT[i][d][s]
// ---------------------------------------------------------------------------
__launch_bounds__(256)
__global__ void k_norm_proj(const float* __restrict__ x,
                            const float* __restrict__ mask,
                            const float* __restrict__ nw,
                            const float* __restrict__ nb,
                            const float* __restrict__ aw,
                            const float* __restrict__ bw,
                            _Float16* __restrict__ aT,
                            _Float16* __restrict__ bT) {
  __shared__ float mrow[8][CIN];
  const int wave = threadIdx.x >> 5;
  const int lane = threadIdx.x & 31;
  const int row  = blockIdx.x * 8 + wave;        // 0 .. S*I-1
  const int s    = row >> 9;                     // row / 512
  const int i    = row & (I_DIM - 1);

  const float* xr = x + (size_t)row * CIN;
  float x0 = xr[lane];
  float x1 = xr[lane + 32];
  float sum = x0 + x1;
  float sq  = x0 * x0 + x1 * x1;
#pragma unroll
  for (int m = 16; m > 0; m >>= 1) {             // wave32 butterfly reduction
    sum += __shfl_xor(sum, m, 32);
    sq  += __shfl_xor(sq,  m, 32);
  }
  const float mu  = sum * (1.0f / CIN);
  const float var = sq * (1.0f / CIN) - mu * mu;
  const float rs  = rsqrtf(var + 1e-5f);

  mrow[wave][lane]      = (x0 - mu) * rs * nw[lane]      + nb[lane];
  mrow[wave][lane + 32] = (x1 - mu) * rs * nw[lane + 32] + nb[lane + 32];
  __syncthreads();

  float av = 0.f, bv = 0.f;
#pragma unroll 8
  for (int k = 0; k < CIN; ++k) {
    const float mk = mrow[wave][k];
    av += mk * aw[k * C_DIM + lane];
    bv += mk * bw[k * D_DIM + lane];
  }
  const float msk = mask[s * I_DIM + i];
  av *= msk * (1.0f / (float)S_DIM);             // fold mask + 1/n_msa into a

  aT[((size_t)i * C_DIM + lane) * S_DIM + s] = (_Float16)av;
  bT[((size_t)i * D_DIM + lane) * S_DIM + s] = (_Float16)bv;
}

// ---------------------------------------------------------------------------
// Kernel 2: fused outer-product-mean + output projection, 16i x 16j per block.
// 512 threads = 16 waves. Loop over 4 c-chunks:
//   Phase A (GEMM1, K=S=128): z[(ii,cc),(jl,d)] into LDS zbuf, B from LDS b-stage
//   Phase B (GEMM2, K=256/chunk): acc[(ii,jl), z] += zbuf @ oWT-slice (regs)
// Epilogue: add bias, store f32 out tile [16,16,128].
// ---------------------------------------------------------------------------
__launch_bounds__(512)
__global__ void k_opm(const _Float16* __restrict__ aT,
                      const _Float16* __restrict__ bT,
                      const _Float16* __restrict__ oWT,
                      const float* __restrict__ obias,
                      float* __restrict__ out) {
  __shared__ __align__(32) _Float16 zbuf[ZROWS * ZS];   // 139,264 B
  __shared__ __align__(32) _Float16 bsh [BROWS * BS];   // 147,456 B (287 KB total)

  const int wave  = threadIdx.x >> 5;            // 0..15
  const int lane  = threadIdx.x & 31;
  const int lhalf = lane >> 4;                   // K-half per WMMA fragment layout
  const int lmod  = lane & 15;
  const int i0 = blockIdx.x * IT;
  const int j0 = blockIdx.y * JT;

  // ---- stage b-tile [16j][32d][128s] -> LDS (read once from L2 per block) ----
  {
    const int r  = threadIdx.x;                  // row = jl*32 + d, 512 rows
    const int jl = r >> 5, d = r & 31;
    const _Float16* src = bT + ((size_t)(j0 + jl) * D_DIM + d) * S_DIM;
    _Float16* dst = bsh + r * BS;
#pragma unroll
    for (int k = 0; k < S_DIM; k += 16)
      *(v16h*)(dst + k) = *(const v16h*)(src + k);
  }

  v8f acc[8];
#pragma unroll
  for (int nt = 0; nt < 8; ++nt) acc[nt] = (v8f){};

  __syncthreads();

  for (int ch = 0; ch < NCHUNK; ++ch) {
    const int c0 = ch * CC;

    // ---------------- Phase A: GEMM1 z-chunk into LDS -----------------------
    // M = (ii,cc): 128 rows -> 8 mtiles; N = (jl,d): 512 -> 32 ntiles.
    // wave w: mtile = w>>1, ntiles (w&1)*16 .. +15.
    {
      const int mtile = wave >> 1;
      const int nbase = (wave & 1) * 16;
      const int arow  = mtile * 16 + lmod;       // M row for A fragment
      const int ii    = arow >> 3, cc = arow & 7;
      const _Float16* Ap =
          aT + ((size_t)(i0 + ii) * C_DIM + c0 + cc) * S_DIM + lhalf * 16;
#pragma unroll 4
      for (int t = 0; t < 16; ++t) {
        const int ntile = nbase + t;
        const int n = ntile * 16 + lmod;         // N index = jl*32 + d = bsh row
        const _Float16* Bp = bsh + n * BS + lhalf * 16;
        v8f zacc = {};
#pragma unroll
        for (int k = 0; k < S_DIM; k += 32) {
          v16h af = *(const v16h*)(Ap + k);      // global, WGP$-resident chunk
          v16h bf = *(const v16h*)(Bp + k);      // LDS (ds_load path)
          zacc = __builtin_amdgcn_wmma_f32_16x16x32_f16(
              false, af, false, bf, (short)0, zacc, false, false);
        }
        // D layout: lane l, vgpr r -> M = mtile*16 + r + 8*(l>=16), N = n
        const int njl = n >> 5, nd = n & 31;
#pragma unroll
        for (int r8 = 0; r8 < 8; ++r8) {
          const int mr  = mtile * 16 + r8 + 8 * lhalf;
          const int ii2 = mr >> 3, cc2 = mr & 7;
          zbuf[(ii2 * JT + njl) * ZS + cc2 * D_DIM + nd] = (_Float16)zacc[r8];
        }
      }
    }
    __syncthreads();

    // ---------------- Phase B: GEMM2 accumulate in registers ----------------
    // wave w owns out rows (i = i0+w, all 16 j) x all 128 z (8 ntiles).
    {
      const _Float16* Az = zbuf + (wave * JT + lmod) * ZS + lhalf * 16;
#pragma unroll 2
      for (int kk = 0; kk < ZCOLS; kk += 32) {
        v16h af = *(const v16h*)(Az + kk);       // LDS A fragment (shared by nt)
#pragma unroll
        for (int nt = 0; nt < 8; ++nt) {
          const int zc = nt * 16 + lmod;
          v16h bf = *(const v16h*)(oWT + (size_t)zc * CD + c0 * D_DIM + kk +
                                   lhalf * 16);  // 64KB slice, WGP$-resident
          acc[nt] = __builtin_amdgcn_wmma_f32_16x16x32_f16(
              false, af, false, bf, (short)0, acc[nt], false, false);
        }
      }
    }
    __syncthreads();   // protect zbuf before next chunk overwrites it
  }

  // ---------------- Epilogue: bias + store ----------------------------------
  const int iGlob = i0 + wave;
#pragma unroll
  for (int nt = 0; nt < 8; ++nt) {
    const int zc = nt * 16 + lmod;
    const float bias = obias[zc];
#pragma unroll
    for (int r8 = 0; r8 < 8; ++r8) {
      const int jl = r8 + 8 * lhalf;
      out[((size_t)iGlob * I_DIM + (j0 + jl)) * CZ + zc] = acc[nt][r8] + bias;
    }
  }
}

// ---------------------------------------------------------------------------
extern "C" void kernel_launch(void* const* d_in, const int* in_sizes, int n_in,
                              void* d_out, int out_size, void* d_ws, size_t ws_size,
                              hipStream_t stream) {
  const float* x    = (const float*)d_in[0];
  const float* mask = (const float*)d_in[1];
  const float* nw   = (const float*)d_in[2];
  const float* nb   = (const float*)d_in[3];
  const float* aw   = (const float*)d_in[4];
  const float* bw   = (const float*)d_in[5];
  const float* ow   = (const float*)d_in[6];
  const float* ob   = (const float*)d_in[7];
  float* out = (float*)d_out;

  // Workspace partition (f16): aT 4MB, bT 4MB, oWT 256KB
  _Float16* aT  = (_Float16*)d_ws;
  _Float16* bT  = aT + (size_t)I_DIM * C_DIM * S_DIM;
  _Float16* oWT = bT + (size_t)I_DIM * D_DIM * S_DIM;

  k_prep_ow<<<dim3((CD * CZ + 255) / 256), dim3(256), 0, stream>>>(ow, oWT);
  k_norm_proj<<<dim3((S_DIM * I_DIM) / 8), dim3(256), 0, stream>>>(
      x, mask, nw, nb, aw, bw, aT, bT);
  k_opm<<<dim3(I_DIM / IT, I_DIM / JT), dim3(512), 0, stream>>>(
      aT, bT, oWT, ob, out);

  (void)in_sizes; (void)n_in; (void)out_size; (void)ws_size;
}
